// SparseAttention_73091753443533
// MI455X (gfx1250) — matrix operand order repaired and checked by reading, not compiled, not run on hardware
//
#include <hip/hip_runtime.h>

typedef __bf16 v16bf __attribute__((ext_vector_type(16)));
typedef float  v8f   __attribute__((ext_vector_type(8)));

#define HIDDEN 128
#define HEADS 8
#define DCH 16

// ---------------------------------------------------------------------------
// Pack a (K x N) row-major fp32 weight matrix into bf16 hi/lo WMMA-B fragments.
// Fragment layout: elem g = ((t*32 + lane)*16 + e), t = nt*(K/32) + kt.
// Lane l of tile t holds col n = nt*16+(l&15), rows k = kt*32+(l>>4)*8+(e&7)+(e>>3)*16.
// ---------------------------------------------------------------------------
__global__ void pack_weights(const float* __restrict__ W, __bf16* __restrict__ hi,
                             __bf16* __restrict__ lo, int K, int N) {
    int g = blockIdx.x * 256 + threadIdx.x;
    if (g >= K * N) return;
    int e    = g & 15;
    int lane = (g >> 4) & 31;
    int t    = g >> 9;
    int ktiles = K >> 5;
    int nt = t / ktiles;
    int kt = t % ktiles;
    int n = nt * 16 + (lane & 15);
    int k = kt * 32 + ((lane >> 4) * 8) + (e & 7) + ((e >> 3) * 16);
    float x = W[(size_t)k * N + n];
    __bf16 h = (__bf16)x;
    hi[g] = h;
    lo[g] = (__bf16)(x - (float)h);
}

// ---------------------------------------------------------------------------
// A fragment (16x32 bf16) from a row-major row pointer: lane holds row (l&15);
// elem e<8 -> K=k0+kap+e, e>=8 -> K=k0+16+kap+(e-8), kap=(l>>4)*8.
// ---------------------------------------------------------------------------
__device__ __forceinline__ void load_a_frag(const float* __restrict__ xrow, int k0,
                                            int kap, v16bf& ah, v16bf& al) {
#pragma unroll
    for (int e = 0; e < 8; ++e) {
        float x = xrow[k0 + kap + e];
        __bf16 h = (__bf16)x;
        ah[e] = h;
        al[e] = (__bf16)(x - (float)h);
        float y = xrow[k0 + 16 + kap + e];
        __bf16 h2 = (__bf16)y;
        ah[e + 8] = h2;
        al[e + 8] = (__bf16)(y - (float)h2);
    }
}

__device__ __forceinline__ void load_a_all(const float* __restrict__ xrow, int kap,
                                           v16bf (&ah)[4], v16bf (&al)[4]) {
#pragma unroll
    for (int kt = 0; kt < 4; ++kt) load_a_frag(xrow, kt * 32, kap, ah[kt], al[kt]);
}

// 16x16 output tile over K=128: 4 k-steps x 3 WMMAs on independent accumulators
__device__ __forceinline__ v8f tile_gemm(const v16bf* __restrict__ Whi,
                                         const v16bf* __restrict__ Wlo,
                                         const v16bf (&ah)[4], const v16bf (&al)[4],
                                         int ktiles, int nt, int lane) {
    v8f chh = {}, chl = {}, clh = {};
#pragma unroll
    for (int kt = 0; kt < 4; ++kt) {
        size_t off = (size_t)(nt * ktiles + kt) * 32 + lane;
        v16bf bh = Whi[off];
        v16bf bl = Wlo[off];
        chh = __builtin_amdgcn_wmma_f32_16x16x32_bf16(false, ah[kt], false, bh, (short)0, chh, false, false);
        chl = __builtin_amdgcn_wmma_f32_16x16x32_bf16(false, ah[kt], false, bl, (short)0, chl, false, false);
        clh = __builtin_amdgcn_wmma_f32_16x16x32_bf16(false, al[kt], false, bh, (short)0, clh, false, false);
    }
    return chh + chl + clh;
}

// ---------------------------------------------------------------------------
__global__ void init_zero(unsigned* __restrict__ p, long long n) {
    long long g = (long long)blockIdx.x * blockDim.x + threadIdx.x;
    if (g < n) p[g] = 0u;
}

// ---------------------------------------------------------------------------
// Node projections: q = query @ Wq (N x 128), kv = key @ Wkv (N x 256)
// Job space: rowTile * 24  (jobs 0..7 -> q cols, jobs 8..23 -> kv cols)
// ---------------------------------------------------------------------------
__global__ __launch_bounds__(256) void proj_nodes(
    const float* __restrict__ query, const float* __restrict__ key,
    const v16bf* __restrict__ wq_hi, const v16bf* __restrict__ wq_lo,
    const v16bf* __restrict__ wkv_hi, const v16bf* __restrict__ wkv_lo,
    float* __restrict__ q_ws, float* __restrict__ kv_ws, int nRT)
{
    int gtid = blockIdx.x * 256 + threadIdx.x;
    int wave = gtid >> 5;
    int lane = threadIdx.x & 31;
    int rt   = wave / 24;
    int job  = wave % 24;
    if (rt >= nRT) return;                     // wave-uniform exit
    int r0 = rt * 16;

    const float* X; const v16bf* Whi; const v16bf* Wlo; float* O; int ldo, nt;
    if (job < 8) { X = query; Whi = wq_hi;  Wlo = wq_lo;  O = q_ws;  ldo = 128; nt = job; }
    else         { X = key;   Whi = wkv_hi; Wlo = wkv_lo; O = kv_ws; ldo = 256; nt = job - 8; }

    int ar  = lane & 15;
    int kap = (lane >> 4) * 8;
    const float* xrow = X + (size_t)(r0 + ar) * HIDDEN;

    v16bf ah[4], al[4];
    load_a_all(xrow, kap, ah, al);
    v8f c = tile_gemm(Whi, Wlo, ah, al, 4, nt, lane);

    int mrow = (lane >> 4) * 8;
#pragma unroll
    for (int r = 0; r < 8; ++r)
        O[(size_t)(r0 + mrow + r) * ldo + nt * 16 + ar] = c[r];
}

// ---------------------------------------------------------------------------
// Per-edge fused: b = interaction_repr @ Wb (16 edges x 256 cols in LDS),
// then logits + pair_logits write + atomic segment max.
// Block = 128 threads (4 waves), one block per 16 edges.
// A tile converted to bf16 hi/lo fragments ONCE per block (cooperatively).
// ---------------------------------------------------------------------------
__global__ __launch_bounds__(128) void edge_bias_logits(
    const float* __restrict__ irep,
    const v16bf* __restrict__ wb_hi, const v16bf* __restrict__ wb_lo,
    const float* __restrict__ q_ws, const float* __restrict__ kv_ws,
    const int* __restrict__ qidx, const int* __restrict__ kidx,
    const float* __restrict__ normalizer,
    float* __restrict__ pair_logits, unsigned* __restrict__ segmax_u)
{
    __shared__ __bf16 ahS[4 * 32 * 16];   // A hi fragments: ((kt*32+lane)*16+e), 4KB
    __shared__ __bf16 alS[4 * 32 * 16];   // A lo fragments, 4KB
    __shared__ float  bS[16 * 256];       // b tile (edges x 2*dH), 16KB

    int tid = threadIdx.x;
    int m0  = blockIdx.x * 16;

    // Phase 1: cooperative A-tile conversion, one (kt, lane) fragment per thread.
    {
        int lt  = tid & 31;                 // target lane of the fragment
        int kt  = tid >> 5;                 // 0..3
        int row = lt & 15;
        int kap = (lt >> 4) * 8;
        const float* src = irep + (size_t)(m0 + row) * HIDDEN + kt * 32 + kap;
        v16bf h, l;
#pragma unroll
        for (int e = 0; e < 8; ++e) {
            float x = src[e];
            __bf16 hh = (__bf16)x;
            h[e] = hh;
            l[e] = (__bf16)(x - (float)hh);
            float y = src[16 + e];
            __bf16 h2 = (__bf16)y;
            h[e + 8] = h2;
            l[e + 8] = (__bf16)(y - (float)h2);
        }
        int fo = (kt * 32 + lt) * 16;
        *(v16bf*)&ahS[fo] = h;
        *(v16bf*)&alS[fo] = l;
    }
    __syncthreads();

    int lane = tid & 31;
    int wv   = tid >> 5;
    int ar   = lane & 15;

    // Phase 2: each wave pulls its 8 fragments from LDS (wide ds loads)
    v16bf ah[4], al[4];
#pragma unroll
    for (int kt = 0; kt < 4; ++kt) {
        int fo = (kt * 32 + lane) * 16;
        ah[kt] = *(const v16bf*)&ahS[fo];
        al[kt] = *(const v16bf*)&alS[fo];
    }

    int mrow = (lane >> 4) * 8;
#pragma unroll
    for (int ct = 0; ct < 4; ++ct) {           // 4 col-tiles/wave -> 256 cols/block
        int nt = wv * 4 + ct;
        v8f c = tile_gemm(wb_hi, wb_lo, ah, al, 4, nt, lane);
#pragma unroll
        for (int r = 0; r < 8; ++r)
            bS[(mrow + r) * 256 + nt * 16 + ar] = c[r];
    }
    __syncthreads();

    // Phase 3: logits, thread -> (edge, head)
    int e = tid >> 3;
    int h = tid & 7;
    int m = m0 + e;
    int qi = qidx[m], ki = kidx[m];
    float nrm = normalizer[0];
    nrm = fminf(fmaxf(nrm, 1.0f), 16.0f);

    const float* qrow = q_ws  + (size_t)qi * 128;
    const float* krow = kv_ws + (size_t)ki * 256;   // k = cols [0,128)

    float acc = 0.0f;
#pragma unroll
    for (int d = 0; d < DCH; ++d) {
        int cch  = d * HEADS + h;
        float bm = bS[e * 256 + cch];          // bias_mul
        float ba = bS[e * 256 + 128 + cch];    // bias_add
        acc = fmaf(qrow[cch], fmaf(krow[cch], 1.0f + bm, ba), acc);
    }
    float logit = acc / nrm;
    pair_logits[(size_t)m * HEADS + h] = logit;

    unsigned u = __float_as_uint(logit);
    u = (u & 0x80000000u) ? ~u : (u | 0x80000000u);
    atomicMax(&segmax_u[qi * HEADS + h], u);
}

__device__ __forceinline__ float unmap_max(unsigned m) {
    return (m & 0x80000000u) ? __uint_as_float(m ^ 0x80000000u) : __uint_as_float(~m);
}

// ---------------------------------------------------------------------------
__global__ void seg_expsum(const float* __restrict__ pair_logits,
                           const int* __restrict__ qidx,
                           const unsigned* __restrict__ segmax_u,
                           float* __restrict__ segsum, int total)
{
    int g = blockIdx.x * 256 + threadIdx.x;
    if (g >= total) return;
    int m = g >> 3;
    int h = g & 7;
    int qi = qidx[m];
    float mx = unmap_max(segmax_u[qi * HEADS + h]);
    float e = __expf(pair_logits[g] - mx);
    atomicAdd(&segsum[qi * HEADS + h], e);
}

// ---------------------------------------------------------------------------
__global__ void scatter_agg(const float* __restrict__ pair_logits,
                            const int* __restrict__ qidx, const int* __restrict__ kidx,
                            const unsigned* __restrict__ segmax_u,
                            const float* __restrict__ segsum,
                            const float* __restrict__ kv_ws,
                            float* __restrict__ agg, long long total)
{
    long long g = (long long)blockIdx.x * 256 + threadIdx.x;
    if (g >= total) return;
    int m = (int)(g >> 7);
    int c = (int)(g & 127);
    int h = c & 7;
    int qi = qidx[m], ki = kidx[m];
    float mx  = unmap_max(segmax_u[qi * HEADS + h]);
    float att = __expf(pair_logits[(size_t)m * HEADS + h] - mx) / segsum[qi * HEADS + h];
    float v   = kv_ws[(size_t)ki * 256 + 128 + c];   // v = cols [128,256)
    atomicAdd(&agg[(size_t)qi * 128 + c], att * v);
}

// ---------------------------------------------------------------------------
// result = agg @ Wo (N x 128)
// ---------------------------------------------------------------------------
__global__ __launch_bounds__(256) void out_gemm(
    const float* __restrict__ agg,
    const v16bf* __restrict__ wo_hi, const v16bf* __restrict__ wo_lo,
    float* __restrict__ out, int nRT)
{
    int gtid = blockIdx.x * 256 + threadIdx.x;
    int wave = gtid >> 5;
    int lane = threadIdx.x & 31;
    int rt   = wave >> 3;
    int nt   = wave & 7;
    if (rt >= nRT) return;
    int r0 = rt * 16;

    int ar  = lane & 15;
    int kap = (lane >> 4) * 8;
    const float* xrow = agg + (size_t)(r0 + ar) * 128;

    v16bf ah[4], al[4];
    load_a_all(xrow, kap, ah, al);
    v8f c = tile_gemm(wo_hi, wo_lo, ah, al, 4, nt, lane);

    int mrow = (lane >> 4) * 8;
#pragma unroll
    for (int r = 0; r < 8; ++r)
        out[(size_t)(r0 + mrow + r) * 128 + nt * 16 + ar] = c[r];
}

// ---------------------------------------------------------------------------
extern "C" void kernel_launch(void* const* d_in, const int* in_sizes, int n_in,
                              void* d_out, int out_size, void* d_ws, size_t ws_size,
                              hipStream_t stream) {
    const float* query = (const float*)d_in[0];
    const float* key   = (const float*)d_in[1];
    const float* irep  = (const float*)d_in[2];
    const int*   qidx  = (const int*)d_in[3];
    const int*   kidx  = (const int*)d_in[4];
    const float* Wq    = (const float*)d_in[5];
    const float* Wkv   = (const float*)d_in[6];
    const float* Wb    = (const float*)d_in[7];
    const float* Wo    = (const float*)d_in[8];
    const float* nrm   = (const float*)d_in[9];

    const int N = in_sizes[0] / HIDDEN;   // 50000
    const int M = in_sizes[3];            // 800000

    // ws: q[N*128] | kv[N*256] | segmax_u[N*8] | segsum[N*8] | agg[N*128] | packed bf16 weights
    float*    ws       = (float*)d_ws;
    float*    q_ws     = ws;
    float*    kv_ws    = ws + (size_t)N * 128;
    unsigned* segmax_u = (unsigned*)(ws + (size_t)N * 384);
    float*    segsum   = ws + (size_t)N * 392;
    float*    agg      = ws + (size_t)N * 400;
    __bf16*   pk       = (__bf16*)(ws + (size_t)N * 528);   // 32B-aligned (N*2112 % 32 == 0)
    __bf16* wq_hi  = pk;            __bf16* wq_lo  = pk + 16384;
    __bf16* wkv_hi = pk + 32768;    __bf16* wkv_lo = pk + 65536;
    __bf16* wb_hi  = pk + 98304;    __bf16* wb_lo  = pk + 131072;
    __bf16* wo_hi  = pk + 163840;   __bf16* wo_lo  = pk + 180224;

    float* result      = (float*)d_out;                     // N*128
    float* pair_logits = (float*)d_out + (size_t)N * 128;   // M*8

    // 0) pack weights into bf16 hi/lo fragment layout (tiny, one-time per call)
    pack_weights<<<(128 * 128 + 255) / 256, 256, 0, stream>>>(Wq,  wq_hi,  wq_lo,  128, 128);
    pack_weights<<<(128 * 256 + 255) / 256, 256, 0, stream>>>(Wkv, wkv_hi, wkv_lo, 128, 256);
    pack_weights<<<(128 * 256 + 255) / 256, 256, 0, stream>>>(Wb,  wb_hi,  wb_lo,  128, 256);
    pack_weights<<<(128 * 128 + 255) / 256, 256, 0, stream>>>(Wo,  wo_hi,  wo_lo,  128, 128);

    // 1) zero segmax/segsum/agg (contiguous N*144 words)
    {
        long long n = (long long)N * 144;
        init_zero<<<(unsigned)((n + 255) / 256), 256, 0, stream>>>(segmax_u, n);
    }
    // 2) node projections
    {
        int nRT = (N + 15) / 16;
        long long waves = (long long)nRT * 24;
        proj_nodes<<<(unsigned)((waves + 7) / 8), 256, 0, stream>>>(
            query, key, (const v16bf*)wq_hi, (const v16bf*)wq_lo,
            (const v16bf*)wkv_hi, (const v16bf*)wkv_lo, q_ws, kv_ws, nRT);
    }
    // 3) fused edge bias GEMM + logits + segment max
    {
        unsigned blocks = (unsigned)((M + 15) / 16);
        edge_bias_logits<<<blocks, 128, 0, stream>>>(
            irep, (const v16bf*)wb_hi, (const v16bf*)wb_lo, q_ws, kv_ws,
            qidx, kidx, nrm, pair_logits, segmax_u);
    }
    // 4) segment exp-sum
    {
        int total = M * HEADS;
        seg_expsum<<<(unsigned)((total + 255) / 256), 256, 0, stream>>>(
            pair_logits, qidx, segmax_u, segsum, total);
    }
    // 5) scatter attn*v into agg
    {
        long long total = (long long)M * 128;
        scatter_agg<<<(unsigned)((total + 255) / 256), 256, 0, stream>>>(
            pair_logits, qidx, kidx, segmax_u, segsum, kv_ws, agg, total);
    }
    // 6) output GEMM
    {
        int nRT = (N + 15) / 16;
        long long waves = (long long)nRT * 8;
        out_gemm<<<(unsigned)((waves + 7) / 8), 256, 0, stream>>>(
            agg, (const v16bf*)wo_hi, (const v16bf*)wo_lo, result, nRT);
    }
}
